// PosteriorScoreModel_48782238548508
// MI455X (gfx1250) — matrix operand order repaired and checked by reading, not compiled, not run on hardware
//
#include <hip/hip_runtime.h>
#include <hip/hip_bf16.h>
#include <math.h>

// Problem constants (match reference)
#define BB 32
#define DD 65536
#define HH 512
#define MM 16384
#define VAR_Y 0.01f
#define LOG_A (-6.907755278982137f)   // log(0.001)
#define LOG_B (4.605170185988092f)    // log(100.0)

#define KBLK1 4096   // split-K slice for gemm1 (K = D)
#define KBLK3 2048   // split-K slice for gemm3 (K = D)

typedef __attribute__((ext_vector_type(16))) __bf16 v16bf;
typedef __attribute__((ext_vector_type(8)))  float  v8f;

// ---------------------------------------------------------------------------
// Fragment loaders: read f32 from global, convert to bf16 in registers.
// A-matrix 16-bit 16x32 layout: lane half h = lane>>4 holds
// K = {16h..16h+15} contiguous (matching the B loaders), row = lane&15.
// ---------------------------------------------------------------------------
__device__ __forceinline__ v16bf load_A_f32(const float* __restrict__ src,
                                            int row0, size_t ld, int k0, int lane) {
  const int r    = row0 + (lane & 15);
  const int half = lane >> 4;
  const float* p = src + (size_t)r * ld + (size_t)k0 + 16 * half;
  const float4* q0 = reinterpret_cast<const float4*>(p);        // K: 16h..16h+15
  float4 a0 = q0[0], a1 = q0[1], b0 = q0[2], b1 = q0[3];
  v16bf v;
  v[0]=(__bf16)a0.x; v[1]=(__bf16)a0.y; v[2]=(__bf16)a0.z; v[3]=(__bf16)a0.w;
  v[4]=(__bf16)a1.x; v[5]=(__bf16)a1.y; v[6]=(__bf16)a1.z; v[7]=(__bf16)a1.w;
  v[8]=(__bf16)b0.x; v[9]=(__bf16)b0.y; v[10]=(__bf16)b0.z; v[11]=(__bf16)b0.w;
  v[12]=(__bf16)b1.x; v[13]=(__bf16)b1.y; v[14]=(__bf16)b1.z; v[15]=(__bf16)b1.w;
  return v;
}

// B-matrix 32x16: element (k, n) at src[k*ld + n] (weights stored row-major
// with k the leading index). Lane holds column n = n0 + (lane&15); K-half
// base = 16*(lane>>4); VGPR j packs K = base+2j, base+2j+1.
__device__ __forceinline__ v16bf load_B_strided(const float* __restrict__ src,
                                                int k0, size_t ld, int n0, int lane) {
  const int n  = n0 + (lane & 15);
  const int kb = k0 + ((lane >> 4) << 4);
  const float* p = src + (size_t)kb * ld + n;
  v16bf v;
#pragma unroll
  for (int i = 0; i < 16; ++i) v[i] = (__bf16)p[(size_t)i * ld];
  return v;
}

// B-matrix 32x16 where element (k, n) = src[n*ld + k]  (i.e. B = src^T,
// per-lane 16 contiguous floats -> fully vectorized b128 loads).
__device__ __forceinline__ v16bf load_B_trans(const float* __restrict__ src,
                                              int k0, size_t ld, int n0, int lane) {
  const int n  = n0 + (lane & 15);
  const int kb = k0 + ((lane >> 4) << 4);
  const float4* q = reinterpret_cast<const float4*>(src + (size_t)n * ld + kb);
  float4 a = q[0], b = q[1], c = q[2], d = q[3];
  v16bf v;
  v[0]=(__bf16)a.x; v[1]=(__bf16)a.y; v[2]=(__bf16)a.z; v[3]=(__bf16)a.w;
  v[4]=(__bf16)b.x; v[5]=(__bf16)b.y; v[6]=(__bf16)b.z; v[7]=(__bf16)b.w;
  v[8]=(__bf16)c.x; v[9]=(__bf16)c.y; v[10]=(__bf16)c.z; v[11]=(__bf16)c.w;
  v[12]=(__bf16)d.x; v[13]=(__bf16)d.y; v[14]=(__bf16)d.z; v[15]=(__bf16)d.w;
  return v;
}

__device__ __forceinline__ v8f wmma_bf16(v16bf a, v16bf b, v8f c) {
  return __builtin_amdgcn_wmma_f32_16x16x32_bf16(false, a, false, b, (short)0, c,
                                                 false, false);
}

// ---------------------------------------------------------------------------
// K0: per-batch sigma / var_x
// ---------------------------------------------------------------------------
__global__ void prep_kernel(const float* __restrict__ t,
                            float* __restrict__ sigma, float* __restrict__ varx) {
  int b = threadIdx.x;
  if (b < BB) {
    float s = expf(LOG_A + (LOG_B - LOG_A) * t[b]);
    sigma[b] = s;
    float s2 = s * s;
    varx[b] = s2 / (1.0f + s2);   // SIGMA_X = 1
  }
}

// K1: zero accumulator regions
__global__ void zero_kernel(float* __restrict__ p, int n) {
  for (int i = blockIdx.x * blockDim.x + threadIdx.x; i < n; i += gridDim.x * blockDim.x)
    p[i] = 0.0f;
}

// ---------------------------------------------------------------------------
// K2: pre1 += xt @ W1   (M=32, K=D split-K, N=H)  -> f32 atomics
// grid (H/64, D/KBLK1), block 128 (4 waves, 16 N-cols each, 2 M-tiles/wave)
// ---------------------------------------------------------------------------
__global__ void __launch_bounds__(128)
gemm1_kernel(const float* __restrict__ xt, const float* __restrict__ W1,
             float* __restrict__ pre1) {
  const int lane = threadIdx.x & 31;
  const int wave = threadIdx.x >> 5;
  const int n0 = blockIdx.x * 64 + wave * 16;
  const int k0 = blockIdx.y * KBLK1;
  v8f acc0 = {}, acc1 = {};
  for (int k = k0; k < k0 + KBLK1; k += 32) {
    v16bf bf = load_B_strided(W1, k, HH, n0, lane);
    v16bf a0 = load_A_f32(xt, 0,  DD, k, lane);
    v16bf a1 = load_A_f32(xt, 16, DD, k, lane);
    acc0 = wmma_bf16(a0, bf, acc0);
    acc1 = wmma_bf16(a1, bf, acc1);
  }
  const int n  = n0 + (lane & 15);
  const int mh = (lane >> 4) << 3;
#pragma unroll
  for (int j = 0; j < 8; ++j) {
    atomicAdd(&pre1[(size_t)(mh + j) * HH + n],      acc0[j]);
    atomicAdd(&pre1[(size_t)(16 + mh + j) * HH + n], acc1[j]);
  }
}

// K3: h = tanh(pre1 + b1 + t)
__global__ void act_kernel(const float* __restrict__ pre1, const float* __restrict__ b1,
                           const float* __restrict__ t, float* __restrict__ hmat) {
  int i = blockIdx.x * blockDim.x + threadIdx.x;
  if (i < BB * HH) {
    int b = i / HH, hh = i % HH;
    hmat[i] = tanhf(pre1[i] + b1[hh] + t[b]);
  }
}

// ---------------------------------------------------------------------------
// K4: z = h @ W2 + b2 ; x = xt - sigma*z   (M=32, K=H, N=D column-parallel)
// grid D/64, block 128
// ---------------------------------------------------------------------------
__global__ void __launch_bounds__(128)
gemm2_kernel(const float* __restrict__ hmat, const float* __restrict__ W2,
             const float* __restrict__ b2, const float* __restrict__ xt,
             const float* __restrict__ sigma,
             float* __restrict__ z, float* __restrict__ x) {
  const int lane = threadIdx.x & 31;
  const int wave = threadIdx.x >> 5;
  const int n0 = blockIdx.x * 64 + wave * 16;
  v8f acc0 = {}, acc1 = {};
  for (int k = 0; k < HH; k += 32) {
    v16bf bf = load_B_strided(W2, k, DD, n0, lane);
    v16bf a0 = load_A_f32(hmat, 0,  HH, k, lane);
    v16bf a1 = load_A_f32(hmat, 16, HH, k, lane);
    acc0 = wmma_bf16(a0, bf, acc0);
    acc1 = wmma_bf16(a1, bf, acc1);
  }
  const int n  = n0 + (lane & 15);
  const int mh = (lane >> 4) << 3;
  const float bv = b2[n];
#pragma unroll
  for (int j = 0; j < 8; ++j) {
    int m = mh + j;
    float zv = acc0[j] + bv;
    z[(size_t)m * DD + n] = zv;
    x[(size_t)m * DD + n] = xt[(size_t)m * DD + n] - sigma[m] * zv;
    m = 16 + mh + j;
    zv = acc1[j] + bv;
    z[(size_t)m * DD + n] = zv;
    x[(size_t)m * DD + n] = xt[(size_t)m * DD + n] - sigma[m] * zv;
  }
}

// K5a: duplicate-index multiplicity
__global__ void count_kernel(const int* __restrict__ obs_idx, float* __restrict__ cnt) {
  int m = blockIdx.x * blockDim.x + threadIdx.x;
  if (m < MM) atomicAdd(&cnt[obs_idx[m]], 1.0f);
}

// K5b: S[b, idx[m]] += y_obs[b,m] - x[b, idx[m]]
__global__ void scatter_kernel(const float* __restrict__ y_obs, const float* __restrict__ x,
                               const int* __restrict__ obs_idx, float* __restrict__ S) {
  int i = blockIdx.x * blockDim.x + threadIdx.x;
  if (i < BB * MM) {
    int b = i / MM, m = i % MM;
    int d = obs_idx[m];
    float r = y_obs[i] - x[(size_t)b * DD + d];
    atomicAdd(&S[(size_t)b * DD + d], r);
  }
}

// K6: exact (Sherman–Morrison) posterior solve + A^T, fused:
//     eD[b,d] = S[b,d] / (var_y + cnt[d]*var_x[b])      (in place over S)
__global__ void solve_kernel(float* __restrict__ S, const float* __restrict__ cnt,
                             const float* __restrict__ varx) {
  int i = blockIdx.x * blockDim.x + threadIdx.x;
  if (i < BB * DD) {
    int b = i / DD, d = i % DD;
    S[i] = S[i] / (VAR_Y + cnt[d] * varx[b]);
  }
}

// ---------------------------------------------------------------------------
// K7: ghT[h,b] = sum_d W2[h,d] * eD[b,d]   (A = W2, B = eD^T, split-K over D)
// grid (H/64, D/KBLK3), block 128
// ---------------------------------------------------------------------------
__global__ void __launch_bounds__(128)
gemm3_kernel(const float* __restrict__ W2, const float* __restrict__ eD,
             float* __restrict__ ghT) {
  const int lane = threadIdx.x & 31;
  const int wave = threadIdx.x >> 5;
  const int b0 = (wave & 1) * 16;
  const int h0 = blockIdx.x * 64 + (wave >> 1) * 32;
  const int k0 = blockIdx.y * KBLK3;
  v8f acc0 = {}, acc1 = {};
  for (int k = k0; k < k0 + KBLK3; k += 32) {
    v16bf bf = load_B_trans(eD, k, DD, b0, lane);     // (k=d, n=b) = eD[b*D+d]
    v16bf a0 = load_A_f32(W2, h0,      DD, k, lane);
    v16bf a1 = load_A_f32(W2, h0 + 16, DD, k, lane);
    acc0 = wmma_bf16(a0, bf, acc0);
    acc1 = wmma_bf16(a1, bf, acc1);
  }
  const int n  = b0 + (lane & 15);
  const int mh = (lane >> 4) << 3;
#pragma unroll
  for (int j = 0; j < 8; ++j) {
    atomicAdd(&ghT[(size_t)(h0 + mh + j) * BB + n],      acc0[j]);
    atomicAdd(&ghT[(size_t)(h0 + 16 + mh + j) * BB + n], acc1[j]);
  }
}

// K8: g_u[b,h] = ghT[h,b] * (1 - h^2)
__global__ void gu_kernel(const float* __restrict__ ghT, const float* __restrict__ hmat,
                          float* __restrict__ gu) {
  int i = blockIdx.x * blockDim.x + threadIdx.x;
  if (i < BB * HH) {
    int b = i / HH, hh = i % HH;
    float hv = hmat[i];
    gu[i] = ghT[(size_t)hh * BB + b] * (1.0f - hv * hv);
  }
}

// ---------------------------------------------------------------------------
// K9: g_xt = g_u @ W1^T, fused final:
//     out = z - sigma*eD + sigma^2*g_xt    (M=32, K=H, N=D column-parallel)
// ---------------------------------------------------------------------------
__global__ void __launch_bounds__(128)
gemm4_kernel(const float* __restrict__ gu, const float* __restrict__ W1,
             const float* __restrict__ z, const float* __restrict__ eD,
             const float* __restrict__ sigma, float* __restrict__ out) {
  const int lane = threadIdx.x & 31;
  const int wave = threadIdx.x >> 5;
  const int n0 = blockIdx.x * 64 + wave * 16;
  v8f acc0 = {}, acc1 = {};
  for (int k = 0; k < HH; k += 32) {
    v16bf bf = load_B_trans(W1, k, HH, n0, lane);     // (k=h, n=d) = W1[d*H+h]
    v16bf a0 = load_A_f32(gu, 0,  HH, k, lane);
    v16bf a1 = load_A_f32(gu, 16, HH, k, lane);
    acc0 = wmma_bf16(a0, bf, acc0);
    acc1 = wmma_bf16(a1, bf, acc1);
  }
  const int n  = n0 + (lane & 15);
  const int mh = (lane >> 4) << 3;
#pragma unroll
  for (int j = 0; j < 8; ++j) {
    int m = mh + j;
    float s = sigma[m];
    size_t o = (size_t)m * DD + n;
    out[o] = z[o] - s * eD[o] + s * s * acc0[j];
    m = 16 + mh + j;
    s = sigma[m];
    o = (size_t)m * DD + n;
    out[o] = z[o] - s * eD[o] + s * s * acc1[j];
  }
}

// ---------------------------------------------------------------------------
extern "C" void kernel_launch(void* const* d_in, const int* in_sizes, int n_in,
                              void* d_out, int out_size, void* d_ws, size_t ws_size,
                              hipStream_t stream) {
  const float* xt      = (const float*)d_in[0];
  const float* t       = (const float*)d_in[1];
  const float* W1      = (const float*)d_in[2];
  const float* b1      = (const float*)d_in[3];
  const float* W2      = (const float*)d_in[4];
  const float* b2      = (const float*)d_in[5];
  const float* y_obs   = (const float*)d_in[6];
  const int*   obs_idx = (const int*)d_in[7];
  float* out = (float*)d_out;

  // Workspace layout (floats). Zero-initialized accumulators first.
  float* ws    = (float*)d_ws;
  float* S     = ws;                   // B*D  (scatter sums -> eD in place)
  float* pre1  = S    + (size_t)BB * DD;  // B*H
  float* ghT   = pre1 + (size_t)BB * HH;  // H*B
  float* cnt   = ghT  + (size_t)HH * BB;  // D
  const int nzero = BB * DD + 2 * BB * HH + DD;
  float* hmat  = cnt  + DD;            // B*H
  float* gu    = hmat + (size_t)BB * HH;  // B*H
  float* z     = gu   + (size_t)BB * HH;  // B*D
  float* x     = z    + (size_t)BB * DD;  // B*D
  float* sigma = x    + (size_t)BB * DD;  // B
  float* varx  = sigma + BB;              // B

  prep_kernel<<<1, 32, 0, stream>>>(t, sigma, varx);
  zero_kernel<<<2048, 256, 0, stream>>>(ws, nzero);

  // Forward MLP
  gemm1_kernel<<<dim3(HH / 64, DD / KBLK1), 128, 0, stream>>>(xt, W1, pre1);
  act_kernel<<<(BB * HH + 255) / 256, 256, 0, stream>>>(pre1, b1, t, hmat);
  gemm2_kernel<<<DD / 64, 128, 0, stream>>>(hmat, W2, b2, xt, sigma, z, x);

  // Residual scatter + exact posterior solve (replaces CG, see analysis)
  count_kernel<<<(MM + 255) / 256, 256, 0, stream>>>(obs_idx, cnt);
  scatter_kernel<<<(BB * MM + 255) / 256, 256, 0, stream>>>(y_obs, x, obs_idx, S);
  solve_kernel<<<(BB * DD + 255) / 256, 256, 0, stream>>>(S, cnt, varx);

  // VJP through the denoiser
  gemm3_kernel<<<dim3(HH / 64, DD / KBLK3), 128, 0, stream>>>(W2, S, ghT);
  gu_kernel<<<(BB * HH + 255) / 256, 256, 0, stream>>>(ghT, hmat, gu);
  gemm4_kernel<<<DD / 64, 128, 0, stream>>>(gu, W1, z, S, sigma, out);
}
